// CogGNN_10763188044165
// MI455X (gfx1250) — matrix operand order repaired and checked by reading, not compiled, not run on hardware
//
#include <hip/hip_runtime.h>
#include <math.h>

#define NNODES 100000
#define NEDGES 400000
#define ETOT   (NEDGES + NNODES)
#define INDIM  384
#define PADDIM 3
#define K1     387
#define K1P    416          // padded to 13*32
#define C1     256          // 4 heads * 64
#define K2     256
#define C2     128          // 2 heads * 64
#define HID    64

typedef __attribute__((ext_vector_type(8)))  float  v8f;
typedef __attribute__((ext_vector_type(8)))  __bf16 v8bf;
typedef __attribute__((ext_vector_type(16))) __bf16 v16bf;

// ---------------- helpers ----------------
__device__ __forceinline__ unsigned fkey(float f) {
    unsigned u = __float_as_uint(f);
    return (u & 0x80000000u) ? ~u : (u | 0x80000000u);   // order-preserving map
}
__device__ __forceinline__ float fkeyinv(unsigned k) {
    unsigned u = (k & 0x80000000u) ? (k ^ 0x80000000u) : ~k;
    return __uint_as_float(u);
}
__device__ __forceinline__ float lrelu(float x) { return x > 0.f ? x : 0.2f * x; }

// ---------------- small utility kernels ----------------
__global__ void k_zero(unsigned* p, long long count) {
    long long i = (long long)blockIdx.x * blockDim.x + threadIdx.x;
    if (i < count) p[i] = 0u;
}

__global__ void k_convert_w(const float* __restrict__ w, __bf16* __restrict__ wb,
                            int K, int KP, int C) {
    long long i = (long long)blockIdx.x * blockDim.x + threadIdx.x;
    if (i >= (long long)KP * C) return;
    int k = (int)(i / C);
    wb[i] = (k < K) ? (__bf16)w[i] : (__bf16)0.f;
}

__global__ void k_build_xpad(const float* __restrict__ x, const float* __restrict__ pad,
                             __bf16* __restrict__ xp) {
    long long i = (long long)blockIdx.x * blockDim.x + threadIdx.x;
    if (i >= (long long)NNODES * K1P) return;
    int n = (int)(i / K1P);
    int c = (int)(i % K1P);
    float v = 0.f;
    if (c < INDIM)      v = x[(long long)n * INDIM + c];
    else if (c < K1)    v = pad[c - INDIM];
    xp[i] = (__bf16)v;
}

// ---------------- WMMA bf16 GEMM: out[M,Cc] = A[M,lda(bf16)] * B[lda,Cc(bf16)] ----------------
// 1 block = 256 threads = 8 waves; block owns one 16-row M tile; wave owns NT 16-col N tiles.
template <int NT>
__global__ void k_gemm_bf16(const __bf16* __restrict__ A, int lda,
                            const __bf16* __restrict__ B, int Cc,
                            float* __restrict__ Cout, int kSteps) {
    const int lane   = threadIdx.x & 31;
    const int wave   = threadIdx.x >> 5;
    const int mtile  = blockIdx.x;
    const int row    = mtile * 16 + (lane & 15);
    const int hiHalf = lane >> 4;        // 0: lanes 0-15, 1: lanes 16-31
    const int aOff   = hiHalf * 8;       // ISA 16-bit A layout: +8 K-offset for upper half
    const __bf16* aRow = A + (size_t)row * lda;

    v8f acc[NT];
#pragma unroll
    for (int t = 0; t < NT; ++t)
#pragma unroll
        for (int e = 0; e < 8; ++e) acc[t][e] = 0.f;

    for (int ks = 0; ks < kSteps; ++ks) {
        const int k0 = ks * 32;
        // A fragment: VGPR0-3 -> K=[aOff..aOff+7], VGPR4-7 -> K=[16+aOff..16+aOff+7]
        v8bf lo = *(const v8bf*)(aRow + k0 + aOff);
        v8bf hi = *(const v8bf*)(aRow + k0 + 16 + aOff);
        v16bf a;
#pragma unroll
        for (int i = 0; i < 8; ++i) { a[i] = lo[i]; a[8 + i] = hi[i]; }

        const __bf16* bCol = B + (size_t)(k0 + aOff) * Cc + (lane & 15);  // lane L holds B column N=L&15, K same order as A
#pragma unroll
        for (int t = 0; t < NT; ++t) {
            const int n0 = (wave * NT + t) * 16;
            v16bf b;
#pragma unroll
            for (int i = 0; i < 8; ++i) { b[i] = bCol[(size_t)i * Cc + n0]; b[8 + i] = bCol[(size_t)(16 + i) * Cc + n0]; }
            acc[t] = __builtin_amdgcn_wmma_f32_16x16x32_bf16(
                false, a, false, b, (short)0, acc[t], false, false);
        }
    }
    // C/D layout: lane L, VGPR r -> M = r + 8*hiHalf, N = L&15
    const int mBase = mtile * 16 + hiHalf * 8;
#pragma unroll
    for (int t = 0; t < NT; ++t) {
        const int n = (wave * NT + t) * 16 + (lane & 15);
#pragma unroll
        for (int r = 0; r < 8; ++r)
            Cout[(size_t)(mBase + r) * Cc + n] = acc[t][r];
    }
}

// ---------------- per-node attention logits ----------------
template <int H>
__global__ void k_attn_logits(const float* __restrict__ h, const float* __restrict__ as,
                              const float* __restrict__ ad,
                              float* __restrict__ als, float* __restrict__ ald) {
    long long i = (long long)blockIdx.x * blockDim.x + threadIdx.x;
    if (i >= (long long)NNODES * H) return;
    int n = (int)(i / H), hh = (int)(i % H);
    const float* hp  = h + (size_t)n * (H * HID) + hh * HID;
    const float* asp = as + hh * HID;
    const float* adp = ad + hh * HID;
    float s = 0.f, d = 0.f;
#pragma unroll 8
    for (int c = 0; c < HID; ++c) { float v = hp[c]; s = fmaf(v, asp[c], s); d = fmaf(v, adp[c], d); }
    als[i] = s; ald[i] = d;
}

// ---------------- edge softmax passes ----------------
template <int H>
__global__ void k_edge_max(const int* __restrict__ ei, const float* __restrict__ als,
                           const float* __restrict__ ald, unsigned* __restrict__ mkey) {
    long long i = (long long)blockIdx.x * blockDim.x + threadIdx.x;
    if (i >= (long long)ETOT * H) return;
    int e = (int)(i / H), hh = (int)(i % H);
    int s = (e < NEDGES) ? ei[e]          : e - NEDGES;   // self loops appended
    int d = (e < NEDGES) ? ei[NEDGES + e] : e - NEDGES;
    float lg = lrelu(als[s * H + hh] + ald[d * H + hh]);
    atomicMax(&mkey[d * H + hh], fkey(lg));
}

template <int H>
__global__ void k_edge_scatter(const int* __restrict__ ei, const float* __restrict__ als,
                               const float* __restrict__ ald, const unsigned* __restrict__ mkey,
                               float* __restrict__ den, const float* __restrict__ h,
                               float* __restrict__ agg) {
    const int PER = H * 8;                   // 8 threads of 8 channels per (edge,head)
    long long i = (long long)blockIdx.x * blockDim.x + threadIdx.x;
    if (i >= (long long)ETOT * PER) return;
    int e   = (int)(i / PER);
    int sub = (int)(i % PER);
    int hh  = sub >> 3;
    int c0  = (sub & 7) * 8;
    int s = (e < NEDGES) ? ei[e]          : e - NEDGES;
    int d = (e < NEDGES) ? ei[NEDGES + e] : e - NEDGES;
    float lg = lrelu(als[s * H + hh] + ald[d * H + hh]);
    float m  = fkeyinv(mkey[d * H + hh]);
    float p  = __expf(lg - m);
    if (c0 == 0) atomicAdd(&den[d * H + hh], p);
    const float* hs = h   + (size_t)s * (H * HID) + hh * HID + c0;
    float*       ag = agg + (size_t)d * (H * HID) + hh * HID + c0;
#pragma unroll
    for (int c = 0; c < 8; ++c) atomicAdd(&ag[c], p * hs[c]);
}

// ---------------- node epilogues ----------------
__global__ void k_node_finish1(const float* __restrict__ agg, const float* __restrict__ den,
                               const float* __restrict__ b1, __bf16* __restrict__ aOut) {
    long long i = (long long)blockIdx.x * blockDim.x + threadIdx.x;
    if (i >= (long long)NNODES * C1) return;
    int n = (int)(i >> 8), c = (int)(i & 255), hh = c >> 6;
    float v = agg[i] / (den[n * 4 + hh] + 1e-16f) + b1[c];
    v = v > 0.f ? v : (__expf(v) - 1.f);   // ELU
    aOut[i] = (__bf16)v;                   // bf16 A matrix for layer-2 WMMA GEMM
}

__global__ void k_node_finish2(const float* __restrict__ agg, const float* __restrict__ den,
                               const float* __restrict__ b2, const float* __restrict__ gate,
                               float* __restrict__ out) {
    long long i = (long long)blockIdx.x * blockDim.x + threadIdx.x;
    if (i >= (long long)NNODES * C2) return;
    int n = (int)(i >> 7), c = (int)(i & 127), hh = c >> 6;
    float v = agg[i] / (den[n * 2 + hh] + 1e-16f) + b2[c];
    v = v > 0.f ? v : (__expf(v) - 1.f);   // ELU
    out[i] = v * gate[c & 63];             // tiled gate
}

__global__ void k_gate(const float* __restrict__ pad, const float* __restrict__ gw,
                       const float* __restrict__ gb, float* __restrict__ gate) {
    int j = threadIdx.x;
    if (j >= HID) return;
    float s = gb[j];
#pragma unroll
    for (int k = 0; k < PADDIM; ++k) s = fmaf(pad[k], gw[k * HID + j], s);
    gate[j] = 1.f / (1.f + __expf(-s));
}

// ---------------- attn MLP: one wave32 per node ----------------
__global__ void k_attn_mlp(const float* __restrict__ h, const float* __restrict__ aw1,
                           const float* __restrict__ ab1, const float* __restrict__ aw2,
                           const float* __restrict__ ab2, float* __restrict__ attn) {
    int lane = threadIdx.x & 31;
    int wave = threadIdx.x >> 5;
    int n = blockIdx.x * 8 + wave;
    if (n >= NNODES) return;
    const float* hp = h + (size_t)n * C2;
    const int j0 = lane, j1 = lane + 32;
    float z0 = ab1[j0], z1 = ab1[j1];
    for (int c = 0; c < C2; ++c) {
        float hv = hp[c];
        z0 = fmaf(hv, aw1[c * HID + j0], z0);
        z1 = fmaf(hv, aw1[c * HID + j1], z1);
    }
    z0 = fmaxf(z0, 0.f); z1 = fmaxf(z1, 0.f);
    float part = z0 * aw2[j0] + z1 * aw2[j1];
#pragma unroll
    for (int m = 16; m >= 1; m >>= 1) part += __shfl_xor(part, m, 32);
    if (lane == 0) attn[n] = 1.f / (1.f + __expf(-(part + ab2[0])));
}

// ---------------- launch ----------------
extern "C" void kernel_launch(void* const* d_in, const int* in_sizes, int n_in,
                              void* d_out, int out_size, void* d_ws, size_t ws_size,
                              hipStream_t stream) {
    const float* x   = (const float*)d_in[0];
    const int*   ei  = (const int*)  d_in[1];
    const float* pad = (const float*)d_in[2];
    const float* W1  = (const float*)d_in[3];
    const float* as1 = (const float*)d_in[4];
    const float* ad1 = (const float*)d_in[5];
    const float* b1  = (const float*)d_in[6];
    const float* W2  = (const float*)d_in[7];
    const float* as2 = (const float*)d_in[8];
    const float* ad2 = (const float*)d_in[9];
    const float* b2  = (const float*)d_in[10];
    const float* aw1 = (const float*)d_in[11];
    const float* ab1 = (const float*)d_in[12];
    const float* aw2 = (const float*)d_in[13];
    const float* ab2 = (const float*)d_in[14];
    const float* gw  = (const float*)d_in[15];
    const float* gb  = (const float*)d_in[16];
    (void)in_sizes; (void)n_in; (void)out_size; (void)ws_size;

    float* outH = (float*)d_out;                 // [N, 128]
    float* outA = outH + (size_t)NNODES * C2;    // [N]

    // bump allocator over workspace (~295 MB, buffers reused across layers)
    char* wsb = (char*)d_ws;
    size_t off = 0;
    auto alloc = [&](size_t bytes) -> char* {
        char* p = wsb + off;
        off = (off + bytes + 255) & ~(size_t)255;
        return p;
    };
    __bf16*   w1b  = (__bf16*)  alloc((size_t)K1P * C1 * 2);
    __bf16*   w2b  = (__bf16*)  alloc((size_t)K2 * C2 * 2);
    __bf16*   xpad = (__bf16*)  alloc((size_t)NNODES * K1P * 2); // layer-1 A; reused as layer-2 A
    float*    h1   = (float*)   alloc((size_t)NNODES * C1 * 4);  // layer-1 H; reused as layer-2 H
    float*    als  = (float*)   alloc((size_t)NNODES * 4 * 4);
    float*    ald  = (float*)   alloc((size_t)NNODES * 4 * 4);
    unsigned* mkey = (unsigned*)alloc((size_t)NNODES * 4 * 4);
    float*    den  = (float*)   alloc((size_t)NNODES * 4 * 4);
    float*    agg  = (float*)   alloc((size_t)NNODES * C1 * 4);  // reused for layer 2
    float*    gate = (float*)   alloc(256);

    const int TB = 256;
    auto blk = [](long long t, int b) { return (unsigned)((t + b - 1) / b); };

    // weight conversion + padded bf16 activations
    k_convert_w<<<blk((long long)K1P * C1, TB), TB, 0, stream>>>(W1, w1b, K1, K1P, C1);
    k_convert_w<<<blk((long long)K2 * C2, TB), TB, 0, stream>>>(W2, w2b, K2, K2, C2);
    k_build_xpad<<<blk((long long)NNODES * K1P, TB), TB, 0, stream>>>(x, pad, xpad);

    // ---------- GAT layer 1 (4 heads) ----------
    k_gemm_bf16<2><<<NNODES / 16, TB, 0, stream>>>(xpad, K1P, w1b, C1, h1, K1P / 32);
    k_attn_logits<4><<<blk((long long)NNODES * 4, TB), TB, 0, stream>>>(h1, as1, ad1, als, ald);
    k_zero<<<blk((long long)NNODES * 4, TB), TB, 0, stream>>>(mkey, (long long)NNODES * 4);
    k_zero<<<blk((long long)NNODES * 4, TB), TB, 0, stream>>>((unsigned*)den, (long long)NNODES * 4);
    k_zero<<<blk((long long)NNODES * C1, TB), TB, 0, stream>>>((unsigned*)agg, (long long)NNODES * C1);
    k_edge_max<4><<<blk((long long)ETOT * 4, TB), TB, 0, stream>>>(ei, als, ald, mkey);
    k_edge_scatter<4><<<blk((long long)ETOT * 32, TB), TB, 0, stream>>>(ei, als, ald, mkey, den, h1, agg);
    k_node_finish1<<<blk((long long)NNODES * C1, TB), TB, 0, stream>>>(agg, den, b1, xpad);

    // ---------- GAT layer 2 (2 heads) ----------
    k_gemm_bf16<1><<<NNODES / 16, TB, 0, stream>>>(xpad, K2, w2b, C2, h1, K2 / 32);
    k_attn_logits<2><<<blk((long long)NNODES * 2, TB), TB, 0, stream>>>(h1, as2, ad2, als, ald);
    k_zero<<<blk((long long)NNODES * 2, TB), TB, 0, stream>>>(mkey, (long long)NNODES * 2);
    k_zero<<<blk((long long)NNODES * 2, TB), TB, 0, stream>>>((unsigned*)den, (long long)NNODES * 2);
    k_zero<<<blk((long long)NNODES * C2, TB), TB, 0, stream>>>((unsigned*)agg, (long long)NNODES * C2);
    k_edge_max<2><<<blk((long long)ETOT * 2, TB), TB, 0, stream>>>(ei, als, ald, mkey);
    k_edge_scatter<2><<<blk((long long)ETOT * 16, TB), TB, 0, stream>>>(ei, als, ald, mkey, den, h1, agg);

    // ---------- gate + final outputs ----------
    k_gate<<<1, 64, 0, stream>>>(pad, gw, gb, gate);
    k_node_finish2<<<blk((long long)NNODES * C2, TB), TB, 0, stream>>>(agg, den, b2, gate, outH);
    k_attn_mlp<<<blk(NNODES, 8), TB, 0, stream>>>(outH, aw1, ab1, aw2, ab2, outA);
}
